// FullAttentionLayer_86260123174780
// MI455X (gfx1250) — matrix operand order repaired and checked by reading, not compiled, not run on hardware
//
#include <hip/hip_runtime.h>

// ---------------------------------------------------------------------------
// Fused full attention layer for MI455X (gfx1250, wave32, WMMA bf16).
//   out = softmax_causal((XWq^T+bq)(XWk^T+bk)^T / sqrt(D)) (XWv^T+bv) Wo^T + bo
// All matmuls via v_wmma_f32_16x16x32_bf16 (f32 accumulate).
// Softmax runs in the log2 domain (log2e folded into the Q scale).
// GEMM: 64x64 tile per wave (4x4 WMMA frags) -> 32 FLOP/byte from L2.
// ---------------------------------------------------------------------------

typedef __attribute__((ext_vector_type(16))) __bf16 v16bf;
typedef __attribute__((ext_vector_type(8)))  float  v8f;

#define D_MODEL 1024
#define SEQ     2048
#define BATCH   2
#define HEADS   16
#define HDIM    64
#define MTOT    (BATCH * SEQ)   // 4096 rows in all flattened GEMMs

union BFrag {
  v16bf v;
  uint4 q[2];
  unsigned int u[8];
  unsigned short s[16];
};

// Native bf16 converts (v_cvt_pk_bf16_f32)
__device__ __forceinline__ unsigned int pk2(float a, float b) {
  union { unsigned int u; __bf16 h[2]; } r;
  r.h[0] = (__bf16)a;
  r.h[1] = (__bf16)b;
  return r.u;
}
__device__ __forceinline__ unsigned short bf16bits(float f) {
  union { unsigned short u; __bf16 h; } r;
  r.h = (__bf16)f;
  return r.u;
}

// 16-lane butterfly reductions on the VALU via DPP16 ROW_XMASK (0x160|mask).
template <int MASK>
__device__ __forceinline__ float dpp_xmask(float v) {
  int p = __builtin_amdgcn_update_dpp(0, __float_as_int(v),
                                      0x160 | MASK, 0xf, 0xf, true);
  return __int_as_float(p);
}
__device__ __forceinline__ float half_max(float v) {
  v = fmaxf(v, dpp_xmask<1>(v));
  v = fmaxf(v, dpp_xmask<2>(v));
  v = fmaxf(v, dpp_xmask<4>(v));
  v = fmaxf(v, dpp_xmask<8>(v));
  return v;
}
__device__ __forceinline__ float half_sum(float v) {
  v += dpp_xmask<1>(v);
  v += dpp_xmask<2>(v);
  v += dpp_xmask<4>(v);
  v += dpp_xmask<8>(v);
  return v;
}

// ---------------------------------------------------------------------------
// Elementwise f32 -> bf16 (8 elements / thread, one b128 store)
// ---------------------------------------------------------------------------
__global__ __launch_bounds__(256)
void cvt_f32_bf16(const float* __restrict__ src,
                  unsigned short* __restrict__ dst, int n)
{
  const int i = (blockIdx.x * blockDim.x + threadIdx.x) * 8;
  if (i >= n) return;
  const float4* p = (const float4*)(src + i);
  const float4 a = p[0], b = p[1];
  uint4 o;
  o.x = pk2(a.x, a.y); o.y = pk2(a.z, a.w);
  o.z = pk2(b.x, b.y); o.w = pk2(b.z, b.w);
  *(uint4*)(dst + i) = o;
}

// ---------------------------------------------------------------------------
// GEMM: C[M=4096,N=1024] = A[4096,1024](bf16) * W[1024,1024](bf16)^T,
// + bias, * alpha. One wave = 64x64 tile (4x4 WMMA frags), K step 32.
// A/B fragments each reused 4x per K step -> 32 FLOP/byte.
// MODE 0: bf16 row-major | MODE 1: f32 row-major | MODE 2: bf16 transposed
// ---------------------------------------------------------------------------
template <int MODE>
__global__ __launch_bounds__(256)
void gemm_wmma(const unsigned short* __restrict__ A,
               const unsigned short* __restrict__ W,
               const float* __restrict__ bias,
               float alpha,
               unsigned short* __restrict__ out16,
               float* __restrict__ out32)
{
  const int lane = threadIdx.x & 31;
  const int wid  = blockIdx.x * (blockDim.x >> 5) + (threadIdx.x >> 5);
  const int tn_tiles = D_MODEL / 64;            // 16
  const int tm = wid / tn_tiles;
  const int tn = wid % tn_tiles;
  if (tm >= MTOT / 64) return;
  const int row0 = tm * 64, col0 = tn * 64;
  const int hi = lane >> 4, lo = lane & 15;

  v8f acc[4][4] = {};

  for (int k0 = 0; k0 < D_MODEL; k0 += 32) {
    BFrag a[4], b[4];
    // A fragments: lane holds row (lo), k-chunks {kb..kb+7, kb+16..kb+23}
    #pragma unroll
    for (int rr = 0; rr < 4; ++rr) {
      const int m  = row0 + rr * 16 + lo;
      const int kb = k0 + hi * 8;
      a[rr].q[0] = *(const uint4*)(A + (size_t)m * D_MODEL + kb);
      a[rr].q[1] = *(const uint4*)(A + (size_t)m * D_MODEL + kb + 16);
    }
    // B fragments: column n = W row n; k-chunks {kb..kb+7, kb+16..kb+23}
    #pragma unroll
    for (int cc = 0; cc < 4; ++cc) {
      const int n  = col0 + cc * 16 + lo;
      const int kb = k0 + hi * 8;
      b[cc].q[0] = *(const uint4*)(W + (size_t)n * D_MODEL + kb);
      b[cc].q[1] = *(const uint4*)(W + (size_t)n * D_MODEL + kb + 16);
    }
    #pragma unroll
    for (int rr = 0; rr < 4; ++rr)
      #pragma unroll
      for (int cc = 0; cc < 4; ++cc)
        acc[rr][cc] = __builtin_amdgcn_wmma_f32_16x16x32_bf16(
            false, a[rr].v, false, b[cc].v, (short)0, acc[rr][cc], false, false);
  }

  // Epilogue: bias + alpha, then store per C/D layout
  #pragma unroll
  for (int cc = 0; cc < 4; ++cc) {
    const int n = col0 + cc * 16 + lo;
    const float bv_ = bias[n];
    #pragma unroll
    for (int rr = 0; rr < 4; ++rr) {
      #pragma unroll
      for (int r = 0; r < 8; ++r) {
        const int m = row0 + rr * 16 + r + hi * 8;
        const float val = (acc[rr][cc][r] + bv_) * alpha;
        if (MODE == 0)      out16[(size_t)m * D_MODEL + n] = bf16bits(val);
        else if (MODE == 1) out32[(size_t)m * D_MODEL + n] = val;
        else                out16[(size_t)n * MTOT + m]    = bf16bits(val);
      }
    }
  }
}

// ---------------------------------------------------------------------------
// Flash attention chunk (32 s-columns). MASKC=true only for diagonal chunk.
// ---------------------------------------------------------------------------
struct AttnState {
  v8f o0, o1, o2, o3;
  float mrow[8], lrow[8];
};

template <bool MASKC>
__device__ __forceinline__ void attn_chunk(
    const unsigned short* __restrict__ K,
    const unsigned short* __restrict__ VT,
    unsigned short (&pl)[16][40],
    const BFrag (&qa)[2], AttnState& st,
    int sbase, int qb, int sc, int hc, int hi, int lo)
{
  // S = Q * K^T for two 16-col tiles (4 WMMAs)
  v8f S0 = {}, S1 = {};
  #pragma unroll
  for (int ct = 0; ct < 2; ++ct) {
    #pragma unroll
    for (int kk = 0; kk < 2; ++kk) {
      const int s  = sbase + ct * 16 + lo;
      const int kb = kk * 32 + hi * 8;
      BFrag kbf;
      kbf.q[0] = *(const uint4*)(K + (size_t)s * D_MODEL + hc + kb);
      kbf.q[1] = *(const uint4*)(K + (size_t)s * D_MODEL + hc + kb + 16);
      if (ct == 0)
        S0 = __builtin_amdgcn_wmma_f32_16x16x32_bf16(
            false, qa[kk].v, false, kbf.v, (short)0, S0, false, false);
      else
        S1 = __builtin_amdgcn_wmma_f32_16x16x32_bf16(
            false, qa[kk].v, false, kbf.v, (short)0, S1, false, false);
    }
  }
  // Online softmax (log2 domain) in C-layout; DPP butterflies per row
  #pragma unroll
  for (int r = 0; r < 8; ++r) {
    float a = S0[r], b = S1[r];
    if (MASKC) {
      const int l   = qb * 16 + r + hi * 8;
      const int s0i = sc * 32 + lo;
      if (s0i > l)      a = -1e30f;
      if (s0i + 16 > l) b = -1e30f;
    }
    const float rm = half_max(fmaxf(a, b));
    const float mn = fmaxf(st.mrow[r], rm);
    const float cf = exp2f(st.mrow[r] - mn);
    a = exp2f(a - mn); b = exp2f(b - mn);
    const float rs = half_sum(a + b);
    st.lrow[r] = st.lrow[r] * cf + rs;
    st.mrow[r] = mn;
    st.o0[r] *= cf; st.o1[r] *= cf; st.o2[r] *= cf; st.o3[r] *= cf;
    // spill P (bf16) to LDS row-major for A-fragment relayout
    const int row = r + hi * 8;
    pl[row][lo]      = bf16bits(a);
    pl[row][16 + lo] = bf16bits(b);
  }
  asm volatile("s_wait_dscnt 0" ::: "memory"); // within-wave cross-lane dep
  // P A-fragment from LDS
  BFrag pa;
  {
    const int row = lo, kb = hi * 8;
    pa.q[0] = *(const uint4*)&pl[row][kb];
    pa.q[1] = *(const uint4*)&pl[row][kb + 16];
  }
  // O += P * V : V^T rows are contiguous in s -> clean B-fragments
  #pragma unroll
  for (int t = 0; t < 4; ++t) {
    const int d = hc + t * 16 + lo;
    const unsigned short* vp = VT + (size_t)d * MTOT + sbase + hi * 8;
    BFrag vb;
    vb.q[0] = *(const uint4*)(vp);
    vb.q[1] = *(const uint4*)(vp + 16);
    v8f* op = (t == 0) ? &st.o0 : (t == 1) ? &st.o1 : (t == 2) ? &st.o2 : &st.o3;
    *op = __builtin_amdgcn_wmma_f32_16x16x32_bf16(
        false, pa.v, false, vb.v, (short)0, *op, false, false);
  }
}

// ---------------------------------------------------------------------------
// Causal flash attention. One wave per (batch, head, 16-row query block).
// Q,K bf16 row-major [4096,1024] (1/sqrt(D)*log2e folded into Q).
// VT bf16 [1024,4096] (transposed). Output bf16 [4096,1024].
// ---------------------------------------------------------------------------
__global__ __launch_bounds__(256)
void attn_wmma(const unsigned short* __restrict__ Q,
               const unsigned short* __restrict__ K,
               const unsigned short* __restrict__ VT,
               unsigned short* __restrict__ O)
{
  __shared__ __align__(16) unsigned short plds[8][16][40]; // per-wave P tile
  const int lane = threadIdx.x & 31;
  const int w    = threadIdx.x >> 5;
  const int wid  = blockIdx.x * 8 + w;           // [0, 4096)
  const int n  = wid >> 11;                      // batch
  const int h  = (wid >> 7) & 15;                // head
  const int qb = wid & 127;                      // query 16-row block
  const int hi = lane >> 4, lo = lane & 15;
  const int hc = h * HDIM;
  const int mq = n * SEQ + qb * 16;              // flattened row base

  // Q A-fragments (kept in registers for the whole s loop)
  BFrag qa[2];
  #pragma unroll
  for (int kk = 0; kk < 2; ++kk) {
    const int m  = mq + lo;
    const int kb = kk * 32 + hi * 8;
    qa[kk].q[0] = *(const uint4*)(Q + (size_t)m * D_MODEL + hc + kb);
    qa[kk].q[1] = *(const uint4*)(Q + (size_t)m * D_MODEL + hc + kb + 16);
  }

  AttnState st;
  st.o0 = v8f{}; st.o1 = v8f{}; st.o2 = v8f{}; st.o3 = v8f{};
  #pragma unroll
  for (int r = 0; r < 8; ++r) { st.mrow[r] = -1e30f; st.lrow[r] = 0.f; }

  const int nchunks = qb / 2 + 1;                // causal: s <= l only
  for (int sc = 0; sc < nchunks - 1; ++sc)       // interior: mask-free
    attn_chunk<false>(K, VT, plds[w], qa, st, n * SEQ + sc * 32,
                      qb, sc, hc, hi, lo);
  attn_chunk<true>(K, VT, plds[w], qa, st,       // diagonal chunk: masked
                   n * SEQ + (nchunks - 1) * 32, qb, nchunks - 1, hc, hi, lo);

  // epilogue: divide by softmax denominator, store bf16
  #pragma unroll
  for (int r = 0; r < 8; ++r) {
    const float inv = 1.0f / st.lrow[r];
    const int m = mq + r + hi * 8;
    O[(size_t)m * D_MODEL + hc +  0 + lo] = bf16bits(st.o0[r] * inv);
    O[(size_t)m * D_MODEL + hc + 16 + lo] = bf16bits(st.o1[r] * inv);
    O[(size_t)m * D_MODEL + hc + 32 + lo] = bf16bits(st.o2[r] * inv);
    O[(size_t)m * D_MODEL + hc + 48 + lo] = bf16bits(st.o3[r] * inv);
  }
}

// ---------------------------------------------------------------------------
extern "C" void kernel_launch(void* const* d_in, const int* in_sizes, int n_in,
                              void* d_out, int out_size, void* d_ws, size_t ws_size,
                              hipStream_t stream)
{
  (void)in_sizes; (void)n_in; (void)out_size; (void)ws_size;
  const float* queries = (const float*)d_in[0];
  const float* keys    = (const float*)d_in[1];
  const float* values  = (const float*)d_in[2];
  const float* Wq = (const float*)d_in[3];
  const float* bq = (const float*)d_in[4];
  const float* Wk = (const float*)d_in[5];
  const float* bk = (const float*)d_in[6];
  const float* Wv = (const float*)d_in[7];
  const float* bv = (const float*)d_in[8];
  const float* Wo = (const float*)d_in[9];
  const float* bo = (const float*)d_in[10];
  float* out = (float*)d_out;

  // Workspace (bf16 elems): q|k|vT|attn (4x4M) + X copies (3x4M) + W copies
  // (4x1M) = 32M elems = 64 MB
  const size_t NX = (size_t)MTOT * D_MODEL;     // 4 Mi elems
  const size_t NW = (size_t)D_MODEL * D_MODEL;  // 1 Mi elems
  unsigned short* qws  = (unsigned short*)d_ws;
  unsigned short* kws  = qws  + NX;
  unsigned short* vtws = kws  + NX;
  unsigned short* aws  = vtws + NX;
  unsigned short* xq   = aws  + NX;
  unsigned short* xk   = xq   + NX;
  unsigned short* xv   = xk   + NX;
  unsigned short* wqb  = xv   + NX;
  unsigned short* wkb  = wqb  + NW;
  unsigned short* wvb  = wkb  + NW;
  unsigned short* wob  = wvb  + NW;

  const dim3 blk(256);
  const dim3 grdX((unsigned)(NX / 8 / 256));            // 2048 blocks
  const dim3 grdW((unsigned)(NW / 8 / 256));            // 512 blocks
  const dim3 grd(((MTOT / 64) * (D_MODEL / 64)) / 8);   // 1024 waves / 8 = 128
  const dim3 grd2((BATCH * HEADS * (SEQ / 16)) / 8);    // 4096 waves / 8 = 512
  // 1/sqrt(HDIM) * log2(e): softmax runs base-2
  const float qscale = 0.125f * 1.44269504088896340736f;

  cvt_f32_bf16<<<grdX, blk, 0, stream>>>(queries, xq, (int)NX);
  cvt_f32_bf16<<<grdX, blk, 0, stream>>>(keys,    xk, (int)NX);
  cvt_f32_bf16<<<grdX, blk, 0, stream>>>(values,  xv, (int)NX);
  cvt_f32_bf16<<<grdW, blk, 0, stream>>>(Wq, wqb, (int)NW);
  cvt_f32_bf16<<<grdW, blk, 0, stream>>>(Wk, wkb, (int)NW);
  cvt_f32_bf16<<<grdW, blk, 0, stream>>>(Wv, wvb, (int)NW);
  cvt_f32_bf16<<<grdW, blk, 0, stream>>>(Wo, wob, (int)NW);

  gemm_wmma<0><<<grd, blk, 0, stream>>>(xq, wqb, bq, qscale, qws,  nullptr);
  gemm_wmma<0><<<grd, blk, 0, stream>>>(xk, wkb, bk, 1.0f,   kws,  nullptr);
  gemm_wmma<2><<<grd, blk, 0, stream>>>(xv, wvb, bv, 1.0f,   vtws, nullptr);
  attn_wmma<<<grd2, blk, 0, stream>>>(qws, kws, vtws, aws);
  gemm_wmma<1><<<grd, blk, 0, stream>>>(aws, wob, bo, 1.0f, nullptr, out);
}
